// EBTAttention_48773648614147
// MI455X (gfx1250) — matrix operand order repaired and checked by reading, not compiled, not run on hardware
//
#include <hip/hip_runtime.h>

// ---------------------------------------------------------------------------
// EBT attention block for MI455X (gfx1250, wave32, WMMA bf16 16x16x32,
// async global->LDS staging tracked with ASYNCcnt)
//   B=2, S=2048, HID=2048, NH=NKV=16, HD=128
// ---------------------------------------------------------------------------

typedef __attribute__((ext_vector_type(16))) __bf16        v16bf;
typedef __attribute__((ext_vector_type(8)))  float         v8f;
typedef __attribute__((ext_vector_type(4)))  unsigned int  v4u;

union FragU { v16bf f; v4u q[2]; };

__device__ __forceinline__ unsigned short f2bf(float x) {
  unsigned int u = __float_as_uint(x);
  u += 0x7FFFu + ((u >> 16) & 1u);        // round-to-nearest-even
  return (unsigned short)(u >> 16);
}

__device__ __forceinline__ v8f wmma_bf16(v16bf a, v16bf b, v8f c) {
  return __builtin_amdgcn_wmma_f32_16x16x32_bf16(
      /*neg_a=*/false, a, /*neg_b=*/false, b,
      /*c_mod=*/(short)0, c, /*reuse_a=*/false, /*reuse_b=*/false);
}

// A-operand fragment from a 32-element K-slice starting at p:
// lane(half) takes K = half*8 + [0,8) and 16 + half*8 + [0,8)
__device__ __forceinline__ v16bf ld_a_frag(const unsigned short* p, int half) {
  FragU fu;
  fu.q[0] = *(const v4u*)(p + half * 8);
  fu.q[1] = *(const v4u*)(p + 16 + half * 8);
  return fu.f;
}

// B-operand fragment from a 32-element K-slice starting at p:
// lane(half) takes K = half*8 + [0,8) and 16 + half*8 + [0,8)
__device__ __forceinline__ v16bf ld_b_frag(const unsigned short* p, int half) {
  FragU fu;
  fu.q[0] = *(const v4u*)(p + half * 8);
  fu.q[1] = *(const v4u*)(p + 16 + half * 8);
  return fu.f;
}

// ---- CDNA5 async global->LDS copy (ASYNCcnt path) -------------------------
// Low 32 bits of a generic pointer to LDS are the LDS byte address
// (ISA 10.2: LDS aperture).
__device__ __forceinline__ unsigned lds_off(const void* p) {
  return (unsigned)(size_t)p;
}

__device__ __forceinline__ void async_ld_b128(unsigned ldsaddr, const void* gaddr) {
  // VDST = per-lane LDS byte address, VADDR = per-lane 64-bit global address
  asm volatile("global_load_async_to_lds_b128 %0, %1, off"
               :: "v"(ldsaddr), "v"(gaddr)
               : "memory");
}

__device__ __forceinline__ void wait_async0() {
  asm volatile("s_wait_asynccnt 0x0" ::: "memory");
}

// ---------------------------------------------------------------------------
// fp32 -> bf16 streaming conversion
// ---------------------------------------------------------------------------
__global__ __launch_bounds__(256) void cvt_f32_bf16(const float* __restrict__ in,
                                                    unsigned short* __restrict__ out,
                                                    long n) {
  long i = (long)blockIdx.x * blockDim.x + threadIdx.x;
  long stride = (long)gridDim.x * blockDim.x;
  for (; i < n; i += stride) out[i] = f2bf(in[i]);
}

// ---------------------------------------------------------------------------
// C[M,N] (f32) = A[M,K] (bf16 row-major) * B[N,K]^T (bf16 row-major)
// Block: 256 threads = 8 waves (2 x 4); wave tile 32x32; block tile 64x128.
// A/B K-slices (64x32 and 128x32) staged in LDS via async copies,
// double-buffered; fragments served by ds_load_b128.
// ---------------------------------------------------------------------------
__global__ __launch_bounds__(256) void gemm_tn_bf16(
    const unsigned short* __restrict__ A,
    const unsigned short* __restrict__ Bm,
    float* __restrict__ C,
    int M, int N, int K)
{
  __shared__ __align__(16) unsigned short abuf[2][64][32];    //  8 KB
  __shared__ __align__(16) unsigned short bbuf[2][128][32];   // 16 KB

  const int t    = threadIdx.x;
  const int lane = t & 31;
  const int wave = t >> 5;
  const int half = lane >> 4;
  const int r    = lane & 15;
  const int wm   = wave >> 2;                  // 0..1
  const int wn   = wave & 3;                   // 0..3
  const int mblk = blockIdx.y * 64;
  const int nblk = blockIdx.x * 128;

  auto stage = [&](int buf, int k0) {
    {  // A tile: 64 rows x 32 bf16 = 256 x 16B chunks, 1 per thread
      const int row = t >> 2, c = (t & 3) * 8;
      async_ld_b128(lds_off(&abuf[buf][row][c]),
                    A + (size_t)(mblk + row) * K + k0 + c);
    }
    #pragma unroll
    for (int i = 0; i < 2; ++i) {  // B tile: 128 rows -> 512 chunks, 2/thread
      const int ch = t + i * 256;
      const int row = ch >> 2, c = (ch & 3) * 8;
      async_ld_b128(lds_off(&bbuf[buf][row][c]),
                    Bm + (size_t)(nblk + row) * K + k0 + c);
    }
  };

  v8f acc00 = {}, acc01 = {}, acc10 = {}, acc11 = {};
  const int nk = K >> 5;

  stage(0, 0);
  for (int it = 0; it < nk; ++it) {
    const int cur = it & 1;
    wait_async0();          // this wave's staged chunks have landed in LDS
    __syncthreads();        // all waves' chunks landed; prev readers done
    if (it + 1 < nk) stage(cur ^ 1, (it + 1) << 5);

    v16bf a0 = ld_a_frag(&abuf[cur][wm * 32 + r][0],      half);
    v16bf a1 = ld_a_frag(&abuf[cur][wm * 32 + 16 + r][0], half);
    v16bf b0 = ld_b_frag(&bbuf[cur][wn * 32 + r][0],      half);
    v16bf b1 = ld_b_frag(&bbuf[cur][wn * 32 + 16 + r][0], half);
    acc00 = wmma_bf16(a0, b0, acc00);
    acc01 = wmma_bf16(a0, b1, acc01);
    acc10 = wmma_bf16(a1, b0, acc10);
    acc11 = wmma_bf16(a1, b1, acc11);
  }

  const int m0 = mblk + wm * 32;
  const int n0 = nblk + wn * 32;
  #pragma unroll
  for (int v = 0; v < 8; ++v) {
    const int mr0 = m0 + v + 8 * half;
    const int mr1 = mr0 + 16;
    C[(size_t)mr0 * N + n0 + r]      = acc00[v];
    C[(size_t)mr0 * N + n0 + 16 + r] = acc01[v];
    C[(size_t)mr1 * N + n0 + r]      = acc10[v];
    C[(size_t)mr1 * N + n0 + 16 + r] = acc11[v];
  }
}

// ---------------------------------------------------------------------------
// RoPE + head split.  qkv: [B*S, 48, 128] f32.
//   Q -> qb [B,16,S,128] bf16,  K -> kb [B,16,S,128] bf16,
//   V -> vt [B,16,128,S] bf16 (transposed: contiguous along j for PV "B")
// ---------------------------------------------------------------------------
__global__ __launch_bounds__(128) void rope_split_kernel(
    const float* __restrict__ qkv,
    const float* __restrict__ cosp,
    const float* __restrict__ sinp,
    unsigned short* __restrict__ qb,
    unsigned short* __restrict__ kb,
    unsigned short* __restrict__ vt)
{
  const int h  = blockIdx.x;          // 0..47
  const int bs = blockIdx.y;          // 0..4095
  const int s  = bs & 2047;
  const int b  = bs >> 11;
  const int d  = threadIdx.x;         // 0..127
  const size_t base = ((size_t)bs * 48 + h) * 128;

  float x = qkv[base + d];
  if (h < 32) {
    const float c     = cosp[s * 128 + d];
    const float sn    = sinp[s * 128 + d];
    const float other = (d < 64) ? -qkv[base + d + 64] : qkv[base + d - 64];
    x = x * c + other * sn;
  }
  const unsigned short v = f2bf(x);
  if (h < 16) {
    qb[(((size_t)b * 16 + h) * 2048 + s) * 128 + d] = v;
  } else if (h < 32) {
    kb[(((size_t)b * 16 + (h - 16)) * 2048 + s) * 128 + d] = v;
  } else {
    vt[(((size_t)b * 16 + (h - 32)) * 128 + d) * 2048 + s] = v;
  }
}

// ---------------------------------------------------------------------------
// Flash attention. grid = (S/128, NH, B), block = 256 = 8 waves.
// All 8 waves share the same (b,h) and KV sequence, so K (32x128) and
// V^T (128x32) tiles are staged ONCE per block via async copies
// (8x global-traffic reduction), double-buffered.
// Per 32-wide KV step: 8 score WMMAs + 8 PV WMMAs per wave.
// ---------------------------------------------------------------------------
__global__ __launch_bounds__(256) void flash_attn_kernel(
    const unsigned short* __restrict__ Qb,
    const unsigned short* __restrict__ Kb,
    const unsigned short* __restrict__ Vt,
    unsigned short* __restrict__ O,      // [B,S,16,128] bf16
    float scale)
{
  __shared__ __align__(16) unsigned short kbuf[2][32][128];   // 16 KB
  __shared__ __align__(16) unsigned short vbuf[2][128][32];   // 16 KB
  __shared__ __align__(16) unsigned short pbuf[8][16][32];    //  8 KB

  const int t    = threadIdx.x;
  const int lane = t & 31;
  const int wave = t >> 5;
  const int half = lane >> 4;
  const int r    = lane & 15;
  const int b    = blockIdx.z;
  const int h    = blockIdx.y;
  const int q0   = blockIdx.x * 128 + wave * 16;

  const unsigned short* Q  = Qb + ((size_t)(b * 16 + h) * 2048) * 128;
  const unsigned short* Kp = Kb + ((size_t)(b * 16 + h) * 2048) * 128;
  const unsigned short* Vp = Vt + ((size_t)(b * 16 + h) * 128) * 2048;

  auto stage = [&](int buf, int j0) {
    #pragma unroll
    for (int i = 0; i < 2; ++i) {  // K tile: 32 rows x 128 bf16 = 512 chunks
      const int ch = t + i * 256;
      const int row = ch >> 4, c = (ch & 15) * 8;
      async_ld_b128(lds_off(&kbuf[buf][row][c]),
                    Kp + (size_t)(j0 + row) * 128 + c);
    }
    #pragma unroll
    for (int i = 0; i < 2; ++i) {  // V tile: 128 d-rows x 32 bf16 = 512 chunks
      const int ch = t + i * 256;
      const int row = ch >> 2, c = (ch & 3) * 8;
      async_ld_b128(lds_off(&vbuf[buf][row][c]),
                    Vp + (size_t)row * 2048 + j0 + c);
    }
  };

  // Q fragments for the full head dim, loaded once (4 x 32 K-steps)
  v16bf qf[4];
  const unsigned short* qrow = Q + (size_t)(q0 + r) * 128;
  #pragma unroll
  for (int kk = 0; kk < 4; ++kk) qf[kk] = ld_a_frag(qrow + kk * 32, half);

  v8f acc[8] = {};
  float mrow[8], lrow[8];
  #pragma unroll
  for (int v = 0; v < 8; ++v) { mrow[v] = -3.0e38f; lrow[v] = 0.0f; }

  stage(0, 0);
  for (int it = 0; it < 64; ++it) {
    const int j0  = it * 32;
    const int cur = it & 1;
    wait_async0();
    __syncthreads();
    if (it + 1 < 64) stage(cur ^ 1, j0 + 32);

    // ---- scores: S = Q(16x128) @ K_tile^T, K from LDS ----
    v8f s0 = {}, s1 = {};
    #pragma unroll
    for (int kk = 0; kk < 4; ++kk) {
      v16bf kf0 = ld_b_frag(&kbuf[cur][r][kk * 32],      half);
      v16bf kf1 = ld_b_frag(&kbuf[cur][16 + r][kk * 32], half);
      s0 = wmma_bf16(qf[kk], kf0, s0);
      s1 = wmma_bf16(qf[kk], kf1, s1);
    }

    // ---- online softmax (row reductions across the 16 column-lanes) ----
    float p0[8], p1[8];
    #pragma unroll
    for (int v = 0; v < 8; ++v) {
      float x0 = s0[v] * scale;
      float x1 = s1[v] * scale;
      float tm = fmaxf(x0, x1);
      #pragma unroll
      for (int m = 1; m < 16; m <<= 1) tm = fmaxf(tm, __shfl_xor(tm, m, 32));
      const float mnew = fmaxf(mrow[v], tm);
      const float al   = __expf(mrow[v] - mnew);
      const float e0   = __expf(x0 - mnew);
      const float e1   = __expf(x1 - mnew);
      float rs = e0 + e1;
      #pragma unroll
      for (int m = 1; m < 16; m <<= 1) rs += __shfl_xor(rs, m, 32);
      lrow[v] = lrow[v] * al + rs;
      mrow[v] = mnew;
      p0[v] = e0;
      p1[v] = e1;
      #pragma unroll
      for (int n = 0; n < 8; ++n) acc[n][v] *= al;
    }

    // ---- transpose P from C-layout to A-layout through LDS ----
    #pragma unroll
    for (int v = 0; v < 8; ++v) {
      pbuf[wave][v + 8 * half][r]      = f2bf(p0[v]);
      pbuf[wave][v + 8 * half][r + 16] = f2bf(p1[v]);
    }
    __syncthreads();
    FragU pf;
    pf.q[0] = *(const v4u*)&pbuf[wave][r][half * 8];
    pf.q[1] = *(const v4u*)&pbuf[wave][r][16 + half * 8];
    __syncthreads();

    // ---- O += P(16x32) @ V(32x128), V from LDS ----
    #pragma unroll
    for (int n = 0; n < 8; ++n) {
      v16bf vf = ld_b_frag(&vbuf[cur][n * 16 + r][0], half);
      acc[n] = wmma_bf16(pf.f, vf, acc[n]);
    }
  }

  // ---- epilogue: divide by row sum, store bf16 [B,S,NH,HD] ----
  #pragma unroll
  for (int v = 0; v < 8; ++v) {
    const float inv = 1.0f / lrow[v];
    const int   row = q0 + v + 8 * half;
    unsigned short* orow = O + (((size_t)b * 2048 + row) * 16 + h) * 128;
    #pragma unroll
    for (int n = 0; n < 8; ++n) orow[n * 16 + r] = f2bf(acc[n][v] * inv);
  }
}

// ---------------------------------------------------------------------------
// Host-side orchestration
// ---------------------------------------------------------------------------
extern "C" void kernel_launch(void* const* d_in, const int* in_sizes, int n_in,
                              void* d_out, int out_size, void* d_ws, size_t ws_size,
                              hipStream_t stream) {
  (void)in_sizes; (void)n_in; (void)out_size; (void)ws_size;

  const float* hidden = (const float*)d_in[0];   // [2,2048,2048]
  const float* cosp   = (const float*)d_in[1];   // [2048,128]
  const float* sinp   = (const float*)d_in[2];   // [2048,128]
  const float* w_qkv  = (const float*)d_in[3];   // [6144,2048]
  const float* w_o    = (const float*)d_in[4];   // [2048,2048]
  float* out = (float*)d_out;                    // [2,2048,2048]

  char* ws = (char*)d_ws;
  size_t off = 0;
  auto alloc = [&](size_t bytes) -> void* {
    void* p = ws + off;
    off += (bytes + 255) & ~(size_t)255;
    return p;
  };

  unsigned short* hid_bf  = (unsigned short*)alloc((size_t)4096 * 2048 * 2);
  unsigned short* wqkv_bf = (unsigned short*)alloc((size_t)6144 * 2048 * 2);
  unsigned short* wo_bf   = (unsigned short*)alloc((size_t)2048 * 2048 * 2);
  float*          qkv     = (float*)         alloc((size_t)4096 * 6144 * 4);
  unsigned short* qb      = (unsigned short*)alloc((size_t)2 * 16 * 2048 * 128 * 2);
  unsigned short* kb      = (unsigned short*)alloc((size_t)2 * 16 * 2048 * 128 * 2);
  unsigned short* vt      = (unsigned short*)alloc((size_t)2 * 16 * 128 * 2048 * 2);
  unsigned short* attn_bf = (unsigned short*)alloc((size_t)4096 * 2048 * 2);

  const float scale = 0.08838834764831843f;      // 1/sqrt(128)

  cvt_f32_bf16<<<2048, 256, 0, stream>>>(hidden, hid_bf,  4096L * 2048);
  cvt_f32_bf16<<<2048, 256, 0, stream>>>(w_qkv,  wqkv_bf, 6144L * 2048);
  cvt_f32_bf16<<<2048, 256, 0, stream>>>(w_o,    wo_bf,   2048L * 2048);

  gemm_tn_bf16<<<dim3(6144 / 128, 4096 / 64), 256, 0, stream>>>(
      hid_bf, wqkv_bf, qkv, 4096, 6144, 2048);

  rope_split_kernel<<<dim3(48, 4096), 128, 0, stream>>>(qkv, cosp, sinp, qb, kb, vt);

  flash_attn_kernel<<<dim3(2048 / 128, 16, 2), 256, 0, stream>>>(
      qb, kb, vt, attn_bf, scale);

  gemm_tn_bf16<<<dim3(2048 / 128, 4096 / 64), 256, 0, stream>>>(
      attn_bf, wo_bf, out, 4096, 2048, 2048);
}
